// BinaryLinear_66597762892272
// MI455X (gfx1250) — compile-verified, matrix operand order checked by reading, not run on hardware
//
#include <hip/hip_runtime.h>

typedef __attribute__((ext_vector_type(16))) _Float16 v16h;
typedef __attribute__((ext_vector_type(8)))  _Float16 v8h;
typedef __attribute__((ext_vector_type(4)))  _Float16 v4h;
typedef __attribute__((ext_vector_type(8)))  float    v8f;

#define M_DIM 8192
#define N_DIM 4096
#define K_DIM 4096

#define BM 128          // block tile M
#define BN 256          // block tile N
#define BK 32           // K chunk == WMMA K
#define LDK (BK + 8)    // padded LDS row stride (80B rows: 16B-aligned, conflict-free)
#define NTHREADS 256    // 8 waves of 32

// pack sign(f0),sign(f1) as two f16 (+1/-1): v_perm grabs the float sign bytes,
// AND keeps bits 15/31, OR in 1.0h exponent bits. 3 VALU ops per 2 elements.
__device__ __forceinline__ unsigned int sign_pack2(unsigned int b0, unsigned int b1) {
    unsigned int tops = __builtin_amdgcn_perm(b1, b0, 0x07000300u);
    return (tops & 0x80008000u) | 0x3C003C00u;
}

// ---------------- sign(W) -> f16 pack, fused with sum(|W|) ----------------
__global__ void signw_absum_kernel(const float* __restrict__ w, _Float16* __restrict__ wh,
                                   float* __restrict__ out_sum, long n4) {
    __shared__ float red[256];
    float s = 0.0f;
    long idx = (long)blockIdx.x * blockDim.x + threadIdx.x;
    long stride = (long)gridDim.x * blockDim.x;
    const float4* w4 = (const float4*)w;
    for (long i = idx; i < n4; i += stride) {
        float4 v = w4[i];
        s += fabsf(v.x) + fabsf(v.y) + fabsf(v.z) + fabsf(v.w);
        uint2 p;
        p.x = sign_pack2(__float_as_uint(v.x), __float_as_uint(v.y));
        p.y = sign_pack2(__float_as_uint(v.z), __float_as_uint(v.w));
        *(uint2*)(wh + i * 4) = p;
    }
    red[threadIdx.x] = s;
    __syncthreads();
    for (int off = 128; off > 0; off >>= 1) {
        if (threadIdx.x < off) red[threadIdx.x] += red[threadIdx.x + off];
        __syncthreads();
    }
    if (threadIdx.x == 0) atomicAdd(out_sum, red[0]);
}

// ---------------- x -> f16 ----------------
__global__ void cvtx_kernel(const float* __restrict__ x, _Float16* __restrict__ xh, long n4) {
    long idx = (long)blockIdx.x * blockDim.x + threadIdx.x;
    long stride = (long)gridDim.x * blockDim.x;
    const float4* x4 = (const float4*)x;
    for (long i = idx; i < n4; i += stride) {
        float4 v = x4[i];
        v4h p = { (_Float16)v.x, (_Float16)v.y, (_Float16)v.z, (_Float16)v.w };
        *(v4h*)(xh + i * 4) = p;
    }
}

// ---------------- fallback-only: plain sum(|W|) ----------------
__global__ void absum_kernel(const float* __restrict__ w, float* __restrict__ out_sum, long n4) {
    __shared__ float red[256];
    float s = 0.0f;
    long idx = (long)blockIdx.x * blockDim.x + threadIdx.x;
    long stride = (long)gridDim.x * blockDim.x;
    const float4* w4 = (const float4*)w;
    for (long i = idx; i < n4; i += stride) {
        float4 v = w4[i];
        s += fabsf(v.x) + fabsf(v.y) + fabsf(v.z) + fabsf(v.w);
    }
    red[threadIdx.x] = s;
    __syncthreads();
    for (int off = 128; off > 0; off >>= 1) {
        if (threadIdx.x < off) red[threadIdx.x] += red[threadIdx.x + off];
        __syncthreads();
    }
    if (threadIdx.x == 0) atomicAdd(out_sum, red[0]);
}

// ---------------- shared WMMA tile compute ----------------
// Fragment layout per ISA: lane L (r=L&15, h=L>>4) holds k = {8h..8h+7, 16+8h..16+8h+7}.
__device__ __forceinline__ void frag_compute(const _Float16* __restrict__ Abuf,
                                             const _Float16* __restrict__ Bbuf,
                                             int waveM0, int waveN0, int r, int h,
                                             v8f (&acc)[4][4]) {
    v16h afrag[4], bfrag[4];
    #pragma unroll
    for (int tm = 0; tm < 4; ++tm) {
        const _Float16* rp = Abuf + (size_t)(waveM0 + tm * 16 + r) * LDK;
        v8h lo = *(const v8h*)(rp + 8 * h);
        v8h hi = *(const v8h*)(rp + 16 + 8 * h);
        afrag[tm] = __builtin_shufflevector(lo, hi, 0,1,2,3,4,5,6,7,8,9,10,11,12,13,14,15);
    }
    #pragma unroll
    for (int tn = 0; tn < 4; ++tn) {
        const _Float16* rp = Bbuf + (size_t)(waveN0 + tn * 16 + r) * LDK;
        v8h lo = *(const v8h*)(rp + 8 * h);
        v8h hi = *(const v8h*)(rp + 16 + 8 * h);
        bfrag[tn] = __builtin_shufflevector(lo, hi, 0,1,2,3,4,5,6,7,8,9,10,11,12,13,14,15);
    }
    #pragma unroll
    for (int tm = 0; tm < 4; ++tm)
        #pragma unroll
        for (int tn = 0; tn < 4; ++tn)
            acc[tm][tn] = __builtin_amdgcn_wmma_f32_16x16x32_f16(
                false, afrag[tm], false, bfrag[tn],
                (short)0, acc[tm][tn], false, false);
}

__device__ __forceinline__ void epilogue_store(float* __restrict__ out, const v8f (&acc)[4][4],
                                               int mbase, int nbase, int waveM0, int waveN0,
                                               int r, int h, float scale) {
    #pragma unroll
    for (int tm = 0; tm < 4; ++tm) {
        #pragma unroll
        for (int tn = 0; tn < 4; ++tn) {
            const int rowbase = mbase + waveM0 + tm * 16 + 8 * h;
            const int col     = nbase + waveN0 + tn * 16 + r;
            #pragma unroll
            for (int v = 0; v < 8; ++v)
                out[(size_t)(rowbase + v) * N_DIM + col] = acc[tm][tn][v] * scale;
        }
    }
}

// ---------------- fast path: pure-f16 GEMM ----------------
__global__ __launch_bounds__(NTHREADS)
void gemm_f16_kernel(const _Float16* __restrict__ xh, const _Float16* __restrict__ wh,
                     float* __restrict__ out, const float* __restrict__ sum_ptr) {
    __shared__ _Float16 Ash[2][BM][LDK];
    __shared__ _Float16 Bsh[2][BN][LDK];

    const int tid  = threadIdx.x;
    const int lane = tid & 31;
    const int wave = tid >> 5;
    const int wr   = wave >> 2;
    const int wc   = wave & 3;
    const int r    = lane & 15;
    const int h    = lane >> 4;

    const int mbase = blockIdx.y * BM;
    const int nbase = blockIdx.x * BN;
    const int waveM0 = wr * 64;
    const int waveN0 = wc * 64;

    v8f acc[4][4];
    #pragma unroll
    for (int i = 0; i < 4; ++i)
        #pragma unroll
        for (int j = 0; j < 4; ++j)
            acc[i][j] = (v8f){0.f,0.f,0.f,0.f,0.f,0.f,0.f,0.f};

    const int NK = K_DIM / BK;
    v8h aReg[2], bReg[4];

    auto loadRegs = [&](int kb) {
        const int kbase = kb * BK;
        #pragma unroll
        for (int j = 0; j < 2; ++j) {               // A: 128 rows x 4 v8h
            int i = tid + NTHREADS * j;
            int rw = i >> 2, kc = i & 3;
            aReg[j] = *(const v8h*)(xh + (size_t)(mbase + rw) * K_DIM + kbase + kc * 8);
        }
        #pragma unroll
        for (int j = 0; j < 4; ++j) {               // B: 256 rows x 4 v8h
            int i = tid + NTHREADS * j;
            int rw = i >> 2, kc = i & 3;
            bReg[j] = *(const v8h*)(wh + (size_t)(nbase + rw) * K_DIM + kbase + kc * 8);
        }
    };
    auto storeLds = [&](int buf) {
        #pragma unroll
        for (int j = 0; j < 2; ++j) {
            int i = tid + NTHREADS * j;
            int rw = i >> 2, kc = i & 3;
            *(v8h*)&Ash[buf][rw][kc * 8] = aReg[j];
        }
        #pragma unroll
        for (int j = 0; j < 4; ++j) {
            int i = tid + NTHREADS * j;
            int rw = i >> 2, kc = i & 3;
            *(v8h*)&Bsh[buf][rw][kc * 8] = bReg[j];
        }
    };

    loadRegs(0);
    storeLds(0);
    __syncthreads();

    for (int kb = 0; kb < NK; ++kb) {
        const int cur = kb & 1;
        if (kb + 1 < NK) loadRegs(kb + 1);
        frag_compute(&Ash[cur][0][0], &Bsh[cur][0][0], waveM0, waveN0, r, h, acc);
        if (kb + 1 < NK) storeLds(cur ^ 1);
        __syncthreads();
    }

    const float scale = sum_ptr[0] * (1.0f / ((float)N_DIM * (float)K_DIM));
    epilogue_store(out, acc, mbase, nbase, waveM0, waveN0, r, h, scale);
}

// ---------------- fallback: fused fp32->f16 GEMM (small workspace) ----------------
__global__ __launch_bounds__(NTHREADS)
void gemm_fused_kernel(const float* __restrict__ x, const float* __restrict__ w,
                       float* __restrict__ out, const float* __restrict__ sum_ptr) {
    __shared__ _Float16 Ash[2][BM][LDK];
    __shared__ _Float16 Bsh[2][BN][LDK];

    const int tid  = threadIdx.x;
    const int lane = tid & 31;
    const int wave = tid >> 5;
    const int wr   = wave >> 2;
    const int wc   = wave & 3;
    const int r    = lane & 15;
    const int h    = lane >> 4;

    const int mbase = blockIdx.y * BM;
    const int nbase = blockIdx.x * BN;
    const int waveM0 = wr * 64;
    const int waveN0 = wc * 64;

    v8f acc[4][4];
    #pragma unroll
    for (int i = 0; i < 4; ++i)
        #pragma unroll
        for (int j = 0; j < 4; ++j)
            acc[i][j] = (v8f){0.f,0.f,0.f,0.f,0.f,0.f,0.f,0.f};

    const int NK = K_DIM / BK;
    float4 aReg[4], bReg[8];

    auto loadRegs = [&](int kb) {
        const int kbase = kb * BK;
        #pragma unroll
        for (int j = 0; j < 4; ++j) {
            int i = tid + NTHREADS * j;
            int rw = i >> 3, kc = i & 7;
            aReg[j] = *(const float4*)(x + (size_t)(mbase + rw) * K_DIM + kbase + kc * 4);
        }
        #pragma unroll
        for (int j = 0; j < 8; ++j) {
            int i = tid + NTHREADS * j;
            int rw = i >> 3, kc = i & 7;
            bReg[j] = *(const float4*)(w + (size_t)(nbase + rw) * K_DIM + kbase + kc * 4);
        }
    };
    auto storeLds = [&](int buf) {
        #pragma unroll
        for (int j = 0; j < 4; ++j) {
            int i = tid + NTHREADS * j;
            int rw = i >> 3, kc = i & 7;
            v4h p = { (_Float16)aReg[j].x, (_Float16)aReg[j].y,
                      (_Float16)aReg[j].z, (_Float16)aReg[j].w };
            *(v4h*)&Ash[buf][rw][kc * 4] = p;
        }
        #pragma unroll
        for (int j = 0; j < 8; ++j) {
            int i = tid + NTHREADS * j;
            int rw = i >> 3, kc = i & 7;
            uint2 p;
            p.x = sign_pack2(__float_as_uint(bReg[j].x), __float_as_uint(bReg[j].y));
            p.y = sign_pack2(__float_as_uint(bReg[j].z), __float_as_uint(bReg[j].w));
            *(uint2*)&Bsh[buf][rw][kc * 4] = p;
        }
    };

    loadRegs(0);
    storeLds(0);
    __syncthreads();

    for (int kb = 0; kb < NK; ++kb) {
        const int cur = kb & 1;
        if (kb + 1 < NK) loadRegs(kb + 1);
        frag_compute(&Ash[cur][0][0], &Bsh[cur][0][0], waveM0, waveN0, r, h, acc);
        if (kb + 1 < NK) storeLds(cur ^ 1);
        __syncthreads();
    }

    const float scale = sum_ptr[0] * (1.0f / ((float)N_DIM * (float)K_DIM));
    epilogue_store(out, acc, mbase, nbase, waveM0, waveN0, r, h, scale);
}

extern "C" void kernel_launch(void* const* d_in, const int* in_sizes, int n_in,
                              void* d_out, int out_size, void* d_ws, size_t ws_size,
                              hipStream_t stream) {
    const float* x = (const float*)d_in[0];   // [8192, 4096] f32
    const float* w = (const float*)d_in[1];   // [4096, 4096] f32
    float* out = (float*)d_out;               // [8192, 4096] f32
    float* sum = (float*)d_ws;                // 4-byte |W| sum accumulator at ws[0]

    hipMemsetAsync(sum, 0, sizeof(float), stream);

    const size_t xh_elems = (size_t)M_DIM * K_DIM;
    const size_t wh_elems = (size_t)N_DIM * K_DIM;
    const size_t need = 256 + (xh_elems + wh_elems) * sizeof(_Float16);
    dim3 grid(N_DIM / BN, M_DIM / BM);        // (16, 64)

    if (ws_size >= need) {
        _Float16* xh = (_Float16*)((char*)d_ws + 256);
        _Float16* wh = xh + xh_elems;
        signw_absum_kernel<<<2048, 256, 0, stream>>>(w, wh, sum, (long)(wh_elems / 4));
        cvtx_kernel<<<4096, 256, 0, stream>>>(x, xh, (long)(xh_elems / 4));
        gemm_f16_kernel<<<grid, NTHREADS, 0, stream>>>(xh, wh, out, sum);
    } else {
        absum_kernel<<<2048, 256, 0, stream>>>(w, sum, (long)(wh_elems / 4));
        gemm_fused_kernel<<<grid, NTHREADS, 0, stream>>>(x, w, out, sum);
    }
}
